// CausalSelfAttention_48344151883921
// MI455X (gfx1250) — compile-verified, matrix operand order checked by reading, not run on hardware
//
#include <hip/hip_runtime.h>
#include <hip/hip_bf16.h>

// ---------------------------------------------------------------------------
// Causal self-attention forward for MI455X (gfx1250), bf16 WMMA pipeline
// with async global->LDS tile staging and double-buffered LDS.
// B=4, S=2048, D_MODEL=1024, H=16, HEAD_DIM=64
// ---------------------------------------------------------------------------

#define BQ   4
#define SQ   2048
#define DM   1024
#define NH   16
#define HD   64
#define SCALE 0.125f   // 64^-0.5

typedef __attribute__((ext_vector_type(8)))  __bf16 bf16x8;
typedef __attribute__((ext_vector_type(16))) __bf16 bf16x16;
typedef __attribute__((ext_vector_type(8)))  float  v8f;
typedef int i32x4 __attribute__((vector_size(16)));

#define AS1 __attribute__((address_space(1)))
#define AS3 __attribute__((address_space(3)))

#if __has_builtin(__builtin_amdgcn_global_load_async_to_lds_b128)
#define HAVE_ASYNC_LDS 1
#else
#define HAVE_ASYNC_LDS 0
#endif

#define WMMA_BF16(a, b, c) \
  __builtin_amdgcn_wmma_f32_16x16x32_bf16(false, (a), false, (b), (short)0, (c), false, false)

// Copy 16 bytes global -> LDS; async (ASYNCcnt) when the toolchain exposes it.
static __device__ __forceinline__ void cp16(__bf16* ldst, const __bf16* gsrc) {
#if HAVE_ASYNC_LDS
  __builtin_amdgcn_global_load_async_to_lds_b128((AS1 i32x4*)gsrc, (AS3 i32x4*)ldst, 0, 0);
#else
  *(bf16x8*)ldst = *(const bf16x8*)gsrc;
#endif
}

static __device__ __forceinline__ void wait_async() {
#if HAVE_ASYNC_LDS
#if __has_builtin(__builtin_amdgcn_s_wait_asynccnt)
  __builtin_amdgcn_s_wait_asynccnt(0);
#else
  asm volatile("s_wait_asynccnt 0x0" ::: "memory");
#endif
#endif
}

// Build a 16x32-bf16 fragment (per ISA 7.12.2 layout) from two 16B chunks:
// lane half h: lo covers K = h*8..h*8+7, hi covers K = 16+h*8..16+h*8+7.
static __device__ __forceinline__ bf16x16 ld_frag(const __bf16* p0, const __bf16* p1) {
  bf16x8 lo = *(const bf16x8*)p0;
  bf16x8 hi = *(const bf16x8*)p1;
  return __builtin_shufflevector(lo, hi, 0,1,2,3,4,5,6,7,8,9,10,11,12,13,14,15);
}

// ---------------------------------------------------------------------------
// fp32 -> bf16 conversion kernels
// ---------------------------------------------------------------------------
__global__ __launch_bounds__(256) void k_cvt(const float* __restrict__ in,
                                             __bf16* __restrict__ out, int n) {
  int i = blockIdx.x * 256 + threadIdx.x;
  if (i < n) out[i] = (__bf16)in[i];
}

// out[N][K] = in[K][N], 32x32 LDS-tiled so both sides are coalesced.
// grid: (N/32, K/32), block 256 (32 x 8).
__global__ __launch_bounds__(256) void k_cvt_t(const float* __restrict__ in,
                                               __bf16* __restrict__ out, int N, int K) {
  __shared__ float tile[32][33];
  const int tx = threadIdx.x & 31, ty = threadIdx.x >> 5;
  const int n0 = blockIdx.x * 32, k0 = blockIdx.y * 32;
#pragma unroll
  for (int i = 0; i < 32; i += 8)
    tile[ty + i][tx] = in[(size_t)(k0 + ty + i) * N + n0 + tx];
  __syncthreads();
#pragma unroll
  for (int i = 0; i < 32; i += 8)
    out[(size_t)(n0 + ty + i) * K + k0 + tx] = (__bf16)tile[tx][ty + i];
}

// ---------------------------------------------------------------------------
// Shared GEMM tile staging: 128x32 bf16 A and B tiles into LDS (stride 40).
// ---------------------------------------------------------------------------
static __device__ __forceinline__ void stage_ab(
    __bf16* sA, __bf16* sB, const __bf16* __restrict__ A,
    const __bf16* __restrict__ Bt, int m0, int n0, int kt, int tid)
{
#pragma unroll
  for (int it = 0; it < 2; ++it) {
    int r = (tid >> 2) + it * 64;
    int c = (tid & 3) << 3;
    cp16(&sA[r * 40 + c], &A [(size_t)(m0 + r) * DM + kt * 32 + c]);
    cp16(&sB[r * 40 + c], &Bt[(size_t)(n0 + r) * DM + kt * 32 + c]);
  }
}

// ---------------------------------------------------------------------------
// QKV projection GEMM: C[8192][3072] = Xbf[8192][1024] * Wqkv_t[3072][1024]^T
// Epilogue scatters Q,K in [bh][s][d] layout and V transposed [bh][d][s], bf16.
// Block = 256 threads (8 waves), tile 128x128, k-step 32, double-buffered LDS.
// ---------------------------------------------------------------------------
__global__ __launch_bounds__(256) void k_gemm_qkv(
    const __bf16* __restrict__ A, const __bf16* __restrict__ Bt,
    const float* __restrict__ bias,
    __bf16* __restrict__ Qo, __bf16* __restrict__ Ko, __bf16* __restrict__ Vto)
{
  __shared__ __bf16 sA[2][128 * 40];
  __shared__ __bf16 sB[2][128 * 40];

  const int tid  = threadIdx.x;
  const int lane = tid & 31, wave = tid >> 5;
  const int wm = wave >> 2, wn = wave & 3;         // 2x4 wave grid
  const int h = lane >> 4, n16 = lane & 15;
  const int m0 = blockIdx.y * 128, n0 = blockIdx.x * 128;

  v8f acc[4][2] = {};

  stage_ab(sA[0], sB[0], A, Bt, m0, n0, 0, tid);
  wait_async();
  __syncthreads();

  for (int kt = 0; kt < 32; ++kt) {
    const int cur = kt & 1;
    if (kt + 1 < 32)
      stage_ab(sA[cur ^ 1], sB[cur ^ 1], A, Bt, m0, n0, kt + 1, tid);

    bf16x16 bfr[2];
#pragma unroll
    for (int nt = 0; nt < 2; ++nt) {
      const __bf16* p = &sB[cur][(wn * 32 + nt * 16 + n16) * 40 + h * 8];
      bfr[nt] = ld_frag(p, p + 16);
    }
#pragma unroll
    for (int mt = 0; mt < 4; ++mt) {
      const __bf16* p = &sA[cur][(wm * 64 + mt * 16 + n16) * 40 + h * 8];
      bf16x16 af = ld_frag(p, p + 16);
      acc[mt][0] = WMMA_BF16(af, bfr[0], acc[mt][0]);
      acc[mt][1] = WMMA_BF16(af, bfr[1], acc[mt][1]);
    }
    wait_async();
    __syncthreads();
  }

  // Epilogue: scatter into head-major Q, K and transposed V (all bf16).
#pragma unroll
  for (int mt = 0; mt < 4; ++mt)
#pragma unroll
    for (int nt = 0; nt < 2; ++nt)
#pragma unroll
      for (int i = 0; i < 8; ++i) {
        int row = m0 + wm * 64 + mt * 16 + i + h * 8;   // token index 0..8191
        int col = n0 + wn * 32 + nt * 16 + n16;         // 0..3071
        float v = acc[mt][nt][i] + bias[col];
        int b = row >> 11, s = row & 2047;
        int part = col >> 10;
        int dmod = col & 1023;
        int hh = dmod >> 6, dd = dmod & 63;
        size_t bh = (size_t)(b * NH + hh);
        __bf16 bv = (__bf16)v;
        if (part == 0)      Qo [(bh * SQ + s) * HD + dd] = bv;
        else if (part == 1) Ko [(bh * SQ + s) * HD + dd] = bv;
        else                Vto[(bh * HD + dd) * SQ + s] = bv;   // V transposed
      }
}

// ---------------------------------------------------------------------------
// Flash-style causal attention.
// Grid: (S/64, B*H). Block: 128 threads = 4 waves; each wave owns 16 queries.
// Computes St = K * Q^T per 64-key tile (softmax rows live per-lane),
// online softmax, O += P^T * V via zero-shuffle P fragment rebuild.
// Double-buffered async K/V tile staging.
// ---------------------------------------------------------------------------
static __device__ __forceinline__ void stage_kv(
    __bf16* sK, __bf16* sV, const __bf16* __restrict__ K,
    const __bf16* __restrict__ Vt, int bh, int s0, int tid)
{
#pragma unroll
  for (int it = 0; it < 4; ++it) {
    int idx = tid + it * 128;
    int r = idx >> 3;
    int c = (idx & 7) << 3;
    cp16(&sK[r * 72 + c], &K [((size_t)bh * SQ + s0 + r) * HD + c]);
    cp16(&sV[r * 72 + c], &Vt[((size_t)bh * HD + r) * SQ + s0 + c]);
  }
}

__global__ __launch_bounds__(128) void k_attn(
    const __bf16* __restrict__ Q, const __bf16* __restrict__ K,
    const __bf16* __restrict__ Vt, __bf16* __restrict__ O)
{
  __shared__ __bf16 sK[2][64 * 72];   // [key][d]
  __shared__ __bf16 sV[2][64 * 72];   // [d][key]  (V transposed)

  const int tid  = threadIdx.x;
  const int lane = tid & 31, wave = tid >> 5;
  const int h = lane >> 4, n16 = lane & 15;
  const int qb = blockIdx.x;             // query tile of 64
  const int bh = blockIdx.y;             // b*16 + head
  const int q_base = qb * 64 + wave * 16;

  // Q^T B-fragments (K-dim = head dim), loaded once from global.
  const __bf16* qrow = Q + ((size_t)bh * SQ + q_base + n16) * HD;
  bf16x16 bq0 = ld_frag(qrow +      h * 8, qrow + 16 + h * 8);
  bf16x16 bq1 = ld_frag(qrow + 32 + h * 8, qrow + 48 + h * 8);

  float m_run = -1e30f, l_run = 0.0f;
  v8f oacc[4] = {};

  stage_kv(sK[0], sV[0], K, Vt, bh, 0, tid);
  wait_async();
  __syncthreads();

  for (int t = 0; t <= qb; ++t) {
    const int cur = t & 1;
    const int s0 = t * 64;
    if (t < qb)
      stage_kv(sK[cur ^ 1], sV[cur ^ 1], K, Vt, bh, s0 + 64, tid);
    const __bf16* cK = sK[cur];
    const __bf16* cV = sV[cur];

    // St[key 64][query 16] = K_tile * Q^T  (4 key sub-tiles, 2 k-steps of 32)
    v8f st[4] = {};
#pragma unroll
    for (int mt = 0; mt < 4; ++mt) {
      const __bf16* p = &cK[(mt * 16 + n16) * 72 + h * 8];
      bf16x16 a0 = ld_frag(p,      p + 16);
      bf16x16 a1 = ld_frag(p + 32, p + 48);
      st[mt] = WMMA_BF16(a0, bq0, st[mt]);
      st[mt] = WMMA_BF16(a1, bq1, st[mt]);
    }

    // scale + causal mask (only diagonal tile can mask)
#pragma unroll
    for (int mt = 0; mt < 4; ++mt)
#pragma unroll
      for (int i = 0; i < 8; ++i) {
        float v = st[mt][i] * SCALE;
        if (t == qb) {
          int sk = s0 + mt * 16 + h * 8 + i;
          int sq = q_base + n16;
          if (sk > sq) v = -1e30f;
        }
        st[mt][i] = v;
      }

    // online softmax: query = lane%16; keys split across lane pair (xor 16)
    float mx = -1e30f;
#pragma unroll
    for (int mt = 0; mt < 4; ++mt)
#pragma unroll
      for (int i = 0; i < 8; ++i) mx = fmaxf(mx, st[mt][i]);
    mx = fmaxf(mx, __shfl_xor(mx, 16));
    float m_new = fmaxf(m_run, mx);
    float alpha = __expf(m_run - m_new);
    float rs = 0.0f;
#pragma unroll
    for (int mt = 0; mt < 4; ++mt)
#pragma unroll
      for (int i = 0; i < 8; ++i) {
        float p = __expf(st[mt][i] - m_new);
        st[mt][i] = p;
        rs += p;
      }
    rs += __shfl_xor(rs, 16);
    l_run = l_run * alpha + rs;
    m_run = m_new;

    // rescale O accumulators (per-query factor -> per-VGPR-element via shfl)
#pragma unroll
    for (int i = 0; i < 8; ++i) {
      float fi = __shfl(alpha, h * 8 + i);
#pragma unroll
      for (int nt = 0; nt < 4; ++nt) oacc[nt][i] *= fi;
    }

    // P^T A-fragments: direct per-lane repack of St accumulators (no shuffles)
    bf16x16 p0, p1;
#pragma unroll
    for (int e = 0; e < 8; ++e) {
      p0[e]     = (__bf16)st[0][e];
      p0[e + 8] = (__bf16)st[1][e];
      p1[e]     = (__bf16)st[2][e];
      p1[e + 8] = (__bf16)st[3][e];
    }

    // O[q 16][d 64] += P^T * V
#pragma unroll
    for (int nt = 0; nt < 4; ++nt) {
      const __bf16* vrow = &cV[(nt * 16 + n16) * 72];
      bf16x16 bv0 = ld_frag(vrow +      h * 8, vrow + 16 + h * 8);
      bf16x16 bv1 = ld_frag(vrow + 32 + h * 8, vrow + 48 + h * 8);
      oacc[nt] = WMMA_BF16(p0, bv0, oacc[nt]);
      oacc[nt] = WMMA_BF16(p1, bv1, oacc[nt]);
    }
    wait_async();
    __syncthreads();
  }

  // normalize and write bf16 O in [B,S,D] layout
  float inv = 1.0f / l_run;
  int b = bh >> 4, hh = bh & 15;
#pragma unroll
  for (int nt = 0; nt < 4; ++nt)
#pragma unroll
    for (int i = 0; i < 8; ++i) {
      float fi = __shfl(inv, h * 8 + i);
      int s = q_base + h * 8 + i;
      int dg = hh * HD + nt * 16 + n16;
      O[((size_t)b * SQ + s) * DM + dg] = (__bf16)(oacc[nt][i] * fi);
    }
}

// ---------------------------------------------------------------------------
// Output projection: out[8192][1024] = Obf[8192][1024] * Wout_t[1024][1024]^T + b
// ---------------------------------------------------------------------------
__global__ __launch_bounds__(256) void k_gemm_out(
    const __bf16* __restrict__ A, const __bf16* __restrict__ Bt,
    const float* __restrict__ bias, float* __restrict__ out)
{
  __shared__ __bf16 sA[2][128 * 40];
  __shared__ __bf16 sB[2][128 * 40];

  const int tid  = threadIdx.x;
  const int lane = tid & 31, wave = tid >> 5;
  const int wm = wave >> 2, wn = wave & 3;
  const int h = lane >> 4, n16 = lane & 15;
  const int m0 = blockIdx.y * 128, n0 = blockIdx.x * 128;

  v8f acc[4][2] = {};

  stage_ab(sA[0], sB[0], A, Bt, m0, n0, 0, tid);
  wait_async();
  __syncthreads();

  for (int kt = 0; kt < 32; ++kt) {
    const int cur = kt & 1;
    if (kt + 1 < 32)
      stage_ab(sA[cur ^ 1], sB[cur ^ 1], A, Bt, m0, n0, kt + 1, tid);

    bf16x16 bfr[2];
#pragma unroll
    for (int nt = 0; nt < 2; ++nt) {
      const __bf16* p = &sB[cur][(wn * 32 + nt * 16 + n16) * 40 + h * 8];
      bfr[nt] = ld_frag(p, p + 16);
    }
#pragma unroll
    for (int mt = 0; mt < 4; ++mt) {
      const __bf16* p = &sA[cur][(wm * 64 + mt * 16 + n16) * 40 + h * 8];
      bf16x16 af = ld_frag(p, p + 16);
      acc[mt][0] = WMMA_BF16(af, bfr[0], acc[mt][0]);
      acc[mt][1] = WMMA_BF16(af, bfr[1], acc[mt][1]);
    }
    wait_async();
    __syncthreads();
  }

#pragma unroll
  for (int mt = 0; mt < 4; ++mt)
#pragma unroll
    for (int nt = 0; nt < 2; ++nt)
#pragma unroll
      for (int i = 0; i < 8; ++i) {
        int row = m0 + wm * 64 + mt * 16 + i + h * 8;
        int col = n0 + wn * 32 + nt * 16 + n16;
        out[(size_t)row * DM + col] = acc[mt][nt][i] + bias[col];
      }
}

// ---------------------------------------------------------------------------
// Launch
// ---------------------------------------------------------------------------
extern "C" void kernel_launch(void* const* d_in, const int* in_sizes, int n_in,
                              void* d_out, int out_size, void* d_ws, size_t ws_size,
                              hipStream_t stream) {
  (void)in_sizes; (void)n_in; (void)out_size; (void)ws_size;
  const float* x     = (const float*)d_in[0];   // [4,2048,1024]
  const float* Wqkv  = (const float*)d_in[1];   // [1024,3072]
  const float* bqkv  = (const float*)d_in[2];   // [3072]
  const float* Wout  = (const float*)d_in[3];   // [1024,1024]
  const float* bout  = (const float*)d_in[4];   // [1024]
  float* out = (float*)d_out;

  char* w = (char*)d_ws;
  const size_t TOK = (size_t)BQ * SQ;                 // 8192 tokens
  __bf16* Xbf    = (__bf16*)(w);                      // 8192*1024         (16 MiB)
  __bf16* WqkvT  = (__bf16*)(w + (16u << 20));        // 3072*1024         ( 6 MiB)
  __bf16* WoutT  = (__bf16*)(w + (22u << 20));        // 1024*1024         ( 2 MiB)
  __bf16* Qbf    = (__bf16*)(w + (24u << 20));        // [bh][s][d]        (16 MiB)
  __bf16* Kbf    = (__bf16*)(w + (40u << 20));        // [bh][s][d]        (16 MiB)
  __bf16* Vtbf   = (__bf16*)(w + (56u << 20));        // [bh][d][s]        (16 MiB)
  __bf16* Obf    = (__bf16*)(w + (72u << 20));        // [b,s,D]           (16 MiB)

  // 1) fp32 -> bf16 conversions (weights transposed, LDS-tiled)
  k_cvt  <<<dim3((TOK * DM) / 256), dim3(256), 0, stream>>>(x, Xbf, (int)(TOK * DM));
  k_cvt_t<<<dim3(3 * DM / 32, DM / 32), dim3(256), 0, stream>>>(Wqkv, WqkvT, 3 * DM, DM);
  k_cvt_t<<<dim3(DM / 32, DM / 32), dim3(256), 0, stream>>>(Wout, WoutT, DM, DM);

  // 2) QKV projection + head-layout scatter (V transposed)
  k_gemm_qkv<<<dim3(3 * DM / 128, TOK / 128), dim3(256), 0, stream>>>(
      Xbf, WqkvT, bqkv, Qbf, Kbf, Vtbf);

  // 3) causal flash attention
  k_attn<<<dim3(SQ / 64, BQ * NH), dim3(128), 0, stream>>>(Qbf, Kbf, Vtbf, Obf);

  // 4) output projection (fp32 result + bias)
  k_gemm_out<<<dim3(DM / 128, TOK / 128), dim3(256), 0, stream>>>(
      Obf, WoutT, bout, out);
}
